// GCNLinkPredictor_2516850835926
// MI455X (gfx1250) — matrix operand order repaired and checked by reading, not compiled, not run on hardware
//
#include <hip/hip_runtime.h>
#include <hip/hip_bf16.h>

typedef __attribute__((ext_vector_type(16))) _Float16 v16h;
typedef __attribute__((ext_vector_type(2)))  _Float16 v2h;
typedef __attribute__((ext_vector_type(8)))  float    v8f;

// ---------------------------------------------------------------------------
// Degree kernels
// ---------------------------------------------------------------------------
__global__ void k_deg_init(float* __restrict__ deg, int n) {
    int i = blockIdx.x * blockDim.x + threadIdx.x;
    if (i < n) deg[i] = 1.0f;                 // self-loop contributes 1
}

__global__ void k_deg_accum(const int* __restrict__ dst, float* __restrict__ deg, int E) {
    int e = blockIdx.x * blockDim.x + threadIdx.x;
    if (e < E) atomicAdd(&deg[dst[e]], 1.0f);
}

__global__ void k_rsqrt_inplace(float* __restrict__ d, int n) {
    int i = blockIdx.x * blockDim.x + threadIdx.x;
    if (i < n) d[i] = rsqrtf(d[i]);           // deg >= 1 always
}

// ---------------------------------------------------------------------------
// WMMA GEMM:  Y[M,64] = (RELU? max(X,0):X)[M,K] @ W[K,64]
// One wave computes one 16-row M tile (N=64 -> 4 f32 accumulators).
// W is staged in LDS PRE-SWIZZLED into the exact CDNA5 WMMA B register
// layout (ISA 7.12.2, 16-bit B Kx16): for tile t = kb*4+nt, lane l, the 16
// halves that lane feeds into v_wmma are stored contiguously, so the B fetch
// is one aligned 32-byte LDS read (2x ds_load_b128) per WMMA.
// ---------------------------------------------------------------------------
template <int K, bool RELU>
__global__ __launch_bounds__(256)
void k_gemm_n64(const float* __restrict__ X, const float* __restrict__ W,
                float* __restrict__ Y, int Mtiles) {
    __shared__ v16h lwb[(K / 32) * 4 * 32];   // == K*64 halves, 32B-aligned
    v2h* lw2 = (v2h*)lwb;

    // Stage + swizzle: pair p -> (B reg-half-pair v, lane l, tile t)
    for (int p = threadIdx.x; p < K * 32; p += 256) {
        int v    = p & 7;
        int l    = (p >> 3) & 31;
        int t    = p >> 8;                    // kb*4 + nt
        int kb   = t >> 2, nt = t & 3;
        int half = l >> 4, m16 = l & 15;
        int kk   = kb * 32 + ((v < 4) ? 2 * v : 16 + 2 * (v - 4)) + 8 * half;
        int n    = nt * 16 + m16;
        v2h pk;
        pk.x = (_Float16)W[kk * 64 + n];
        pk.y = (_Float16)W[(kk + 1) * 64 + n];
        lw2[p] = pk;                          // one ds_store_b32
    }
    __syncthreads();

    const int wave = threadIdx.x >> 5;
    const int tile = blockIdx.x * 8 + wave;
    if (tile >= Mtiles) return;               // wave-uniform; EXEC all-1 for WMMA

    const int lane = threadIdx.x & 31;
    const int half = lane >> 4;               // lane group (0: lanes 0-15, 1: 16-31)
    const int m16  = lane & 15;

    const float* xr = X + (size_t)(tile * 16 + m16) * K;

    v8f acc[4] = {};

#pragma unroll
    for (int kb = 0; kb < K / 32; ++kb) {
        const int kbase = kb * 32;
        // ---- A tile: row m16 of X (ISA 16-bit A 16x32 layout); per lane this
        // is two contiguous 8-float runs -> global_load_b128 + v_cvt_pk_f16_f32
        v16h a;
#pragma unroll
        for (int v = 0; v < 8; ++v) {
            int k0 = kbase + ((v < 4) ? 2 * v : 16 + 2 * (v - 4)) + 8 * half;
            float f0 = xr[k0], f1 = xr[k0 + 1];
            if (RELU) { f0 = fmaxf(f0, 0.0f); f1 = fmaxf(f1, 0.0f); }
            a[2 * v]     = (_Float16)f0;
            a[2 * v + 1] = (_Float16)f1;
        }
        // ---- 4 N-tiles: single v16h LDS read each, then WMMA
#pragma unroll
        for (int nt = 0; nt < 4; ++nt) {
            v16h b = lwb[(kb * 4 + nt) * 32 + lane];
            acc[nt] = __builtin_amdgcn_wmma_f32_16x16x32_f16(
                false, a, false, b, (short)0, acc[nt], false, false);
        }
    }

    // ---- store D: VGPR r holds (M = r + 8*half, N = m16)
    float* yr = Y + (size_t)tile * 16 * 64;
#pragma unroll
    for (int nt = 0; nt < 4; ++nt)
#pragma unroll
        for (int r = 0; r < 8; ++r)
            yr[(size_t)(r + 8 * half) * 64 + nt * 16 + m16] = acc[nt][r];
}

// ---------------------------------------------------------------------------
// agg[i,h] = bias[h] + xw[i,h] * dinv[i]^2   (self-loop message + bias)
// ---------------------------------------------------------------------------
__global__ void k_agg_init(const float* __restrict__ xw, const float* __restrict__ dinv,
                           const float* __restrict__ bias, float* __restrict__ agg,
                           int total /* n*64 */) {
    int g = blockIdx.x * blockDim.x + threadIdx.x;
    if (g >= total) return;
    int i = g >> 6, h = g & 63;
    float di = dinv[i];
    agg[g] = bias[h] + xw[g] * di * di;
}

// ---------------------------------------------------------------------------
// Edge scatter: agg[dst] += xw[src] * dinv[src]*dinv[dst]
// 16 threads per edge, float4 gather, 4 f32 atomics each. Working set
// (xw + agg ~51 MB) fits in the 192 MB L2, so atomics coalesce there.
// ---------------------------------------------------------------------------
__global__ __launch_bounds__(256)
void k_scatter(const int* __restrict__ src, const int* __restrict__ dst,
               const float* __restrict__ dinv, const float* __restrict__ xw,
               float* __restrict__ agg, int E) {
    int g = blockIdx.x * blockDim.x + threadIdx.x;
    int e = g >> 4;
    if (e >= E) return;
    int q = (g & 15) * 4;
    int s = src[e], d = dst[e];
    float nrm = dinv[s] * dinv[d];
    const float4 m = *(const float4*)(xw + (size_t)s * 64 + q);
    float* ap = agg + (size_t)d * 64 + q;
    atomicAdd(ap + 0, m.x * nrm);
    atomicAdd(ap + 1, m.y * nrm);
    atomicAdd(ap + 2, m.z * nrm);
    atomicAdd(ap + 3, m.w * nrm);
}

// ---------------------------------------------------------------------------
// Decode: out[e] = sum_h relu(z[s,h])*relu(z[d,h])*lw[h] + lb
// One wave32 per edge; 2 channels per lane; shuffle reduction.
// ---------------------------------------------------------------------------
__global__ __launch_bounds__(256)
void k_decode(const float* __restrict__ z, const int* __restrict__ src,
              const int* __restrict__ dst, const float* __restrict__ lw,
              const float* __restrict__ lb, float* __restrict__ out, int E) {
    int w = (blockIdx.x * blockDim.x + threadIdx.x) >> 5;
    int lane = threadIdx.x & 31;
    if (w >= E) return;
    const float* zs = z + (size_t)src[w] * 64;
    const float* zd = z + (size_t)dst[w] * 64;
    float p = 0.0f;
#pragma unroll
    for (int j = 0; j < 2; ++j) {
        int c = lane + 32 * j;
        p += fmaxf(zs[c], 0.0f) * fmaxf(zd[c], 0.0f) * lw[c];
    }
#pragma unroll
    for (int off = 16; off > 0; off >>= 1)
        p += __shfl_down(p, off, 32);
    if (lane == 0) out[w] = p + lb[0];
}

// ---------------------------------------------------------------------------
extern "C" void kernel_launch(void* const* d_in, const int* in_sizes, int n_in,
                              void* d_out, int out_size, void* d_ws, size_t ws_size,
                              hipStream_t stream) {
    const float* x    = (const float*)d_in[0];
    const int*   ei   = (const int*)d_in[1];
    const int*   pei  = (const int*)d_in[2];
    const int*   nei  = (const int*)d_in[3];
    const float* W1   = (const float*)d_in[4];
    const float* b1   = (const float*)d_in[5];
    const float* W2   = (const float*)d_in[6];
    const float* b2   = (const float*)d_in[7];
    const float* lwv  = (const float*)d_in[8];
    const float* lbv  = (const float*)d_in[9];

    const int n  = in_sizes[0] / 128;   // 100000 nodes
    const int E  = in_sizes[1] / 2;     // 3.2M edges
    const int Ep = in_sizes[2] / 2;     // 500k eval edges each

    const int* src = ei;
    const int* dst = ei + E;

    // workspace layout: dinv | bufA (n*64) | bufB (n*64)
    float* dinv = (float*)d_ws;
    float* bufA = dinv + ((n + 255) & ~255);
    float* bufB = bufA + (size_t)n * 64;

    const int T = 256;
    const int Mtiles  = (n + 15) / 16;           // 6250 (n divisible by 16)
    const int gemmBlk = (Mtiles + 7) / 8;

    // --- degrees + symmetric norm
    k_deg_init<<<(n + T - 1) / T, T, 0, stream>>>(dinv, n);
    k_deg_accum<<<(E + T - 1) / T, T, 0, stream>>>(dst, dinv, E);
    k_rsqrt_inplace<<<(n + T - 1) / T, T, 0, stream>>>(dinv, n);

    // --- layer 1: xw1 = x @ W1 (K=128) -> bufA
    k_gemm_n64<128, false><<<gemmBlk, T, 0, stream>>>(x, W1, bufA, Mtiles);
    // agg1 = b1 + self-loop term, then edge scatter -> bufB
    k_agg_init<<<((n * 64) + T - 1) / T, T, 0, stream>>>(bufA, dinv, b1, bufB, n * 64);
    k_scatter<<<((size_t)E * 16 + T - 1) / T, T, 0, stream>>>(src, dst, dinv, bufA, bufB, E);

    // --- layer 2: xw2 = relu(agg1) @ W2 (K=64) -> bufA (reuse)
    k_gemm_n64<64, true><<<gemmBlk, T, 0, stream>>>(bufB, W2, bufA, Mtiles);
    k_agg_init<<<((n * 64) + T - 1) / T, T, 0, stream>>>(bufA, dinv, b2, bufB, n * 64);
    k_scatter<<<((size_t)E * 16 + T - 1) / T, T, 0, stream>>>(src, dst, dinv, bufA, bufB, E);

    // --- decode (relu folded in): pos then neg, concatenated in d_out
    float* out = (float*)d_out;
    k_decode<<<(Ep + 7) / 8, T, 0, stream>>>(bufB, pei, pei + Ep, lwv, lbv, out, Ep);
    k_decode<<<(Ep + 7) / 8, T, 0, stream>>>(bufB, nei, nei + Ep, lwv, lbv, out + Ep, Ep);
}
